// Cross_TransformerBlock_30923764531523
// MI455X (gfx1250) — compile-verified, matrix operand order checked by reading, not run on hardware
//
#include <hip/hip_runtime.h>
#include <stdint.h>

// ---------------- problem constants ----------------
#define Bb      4
#define HH      128
#define WWd     128
#define C       192
#define HEADS   6
#define G       8
#define N       64          // tokens per window
#define DH      32          // head dim
#define T_TOK   (Bb * HH * WWd)   // 65536 tokens
#define EPSLN   1e-5f

typedef __attribute__((ext_vector_type(16))) __bf16 bf16x16;
typedef __attribute__((ext_vector_type(8)))  float  v8f;

// ---------------- helpers ----------------
__device__ __forceinline__ unsigned short f2bf(float f) {
    union { float f; uint32_t u; } v; v.f = f;
    uint32_t r = (v.u + 0x7FFFu + ((v.u >> 16) & 1u)) >> 16;
    return (unsigned short)r;
}

struct Frag { union { uint32_t u[8]; bf16x16 v; }; };

// Load a 16x32 bf16 A/B fragment from a row-major [rows x stride] bf16 buffer.
// Lane layout per CDNA5 ISA: lanes 0-15 row=row0+lane, K 0-7 & 16-23;
// lanes 16-31 same rows, K 8-15 & 24-31. Two aligned 16B LDS loads per lane.
__device__ __forceinline__ Frag load_frag(const unsigned short* base, int stride, int row0) {
    int lane = threadIdx.x & 31;
    int r  = row0 + (lane & 15);
    int kh = (lane >> 4) << 3;            // 0 or 8 (element offset)
    const uint32_t* p = (const uint32_t*)(base + (size_t)r * stride);
    uint4 a = *(const uint4*)(p + (kh >> 1));
    uint4 b = *(const uint4*)(p + 8 + (kh >> 1));
    Frag f;
    f.u[0] = a.x; f.u[1] = a.y; f.u[2] = a.z; f.u[3] = a.w;
    f.u[4] = b.x; f.u[5] = b.y; f.u[6] = b.z; f.u[7] = b.w;
    return f;
}

__device__ __forceinline__ v8f wmma_bf16(const Frag& a, const Frag& b, v8f c) {
    return __builtin_amdgcn_wmma_f32_16x16x32_bf16(
        /*neg_a=*/false, a.v, /*neg_b=*/false, b.v,
        /*c_mod=*/(short)0, c, /*reuse_a=*/false, /*reuse_b=*/false);
}

// ---- CDNA5 async global->LDS copies (tracked with ASYNCcnt) ----
__device__ __forceinline__ void async_copy_b128(unsigned short* lds, const unsigned short* g) {
    uint32_t l = (uint32_t)(uintptr_t)lds;          // low 32 bits of generic shared ptr = LDS offset
    uint64_t a = (uint64_t)(uintptr_t)g;
    asm volatile("global_load_async_to_lds_b128 %0, %1, off" :: "v"(l), "v"(a) : "memory");
}
__device__ __forceinline__ void async_copy_b32(unsigned short* lds, const unsigned short* g) {
    uint32_t l = (uint32_t)(uintptr_t)lds;
    uint64_t a = (uint64_t)(uintptr_t)g;
    asm volatile("global_load_async_to_lds_b32 %0, %1, off" :: "v"(l), "v"(a) : "memory");
}
__device__ __forceinline__ void wait_async0() {
    asm volatile("s_wait_asynccnt 0x0" ::: "memory");
}

// ---------------- float -> bf16 convert (plain) ----------------
__global__ void cvt_bf16_kernel(const float* __restrict__ in, unsigned short* __restrict__ out, int n) {
    for (int i = blockIdx.x * blockDim.x + threadIdx.x; i < n; i += gridDim.x * blockDim.x)
        out[i] = f2bf(in[i]);
}

// ---------------- float -> bf16 transpose: out[n*K+k] = in[k*Nc+n] ----------------
__global__ void cvt_transpose_kernel(const float* __restrict__ in, unsigned short* __restrict__ out,
                                     int K, int Nc) {
    int total = K * Nc;
    for (int i = blockIdx.x * blockDim.x + threadIdx.x; i < total; i += gridDim.x * blockDim.x) {
        int n = i / K, k = i - n * K;
        out[(size_t)n * K + k] = f2bf(in[(size_t)k * Nc + n]);
    }
}

// ---------------- LayerNorm (f32 in, bf16 out), 1 warp per token ----------------
__global__ __launch_bounds__(256)
void ln_kernel(const float* __restrict__ x, const float* __restrict__ g,
               const float* __restrict__ b, unsigned short* __restrict__ y) {
    int wv = threadIdx.x >> 5, lane = threadIdx.x & 31;
    size_t t = (size_t)blockIdx.x * 8 + wv;
    const float* xr = x + t * C;
    float v[6]; float s = 0.f, sq = 0.f;
#pragma unroll
    for (int i = 0; i < 6; ++i) { v[i] = xr[lane + 32 * i]; s += v[i]; sq += v[i] * v[i]; }
#pragma unroll
    for (int off = 1; off < 32; off <<= 1) { s += __shfl_xor(s, off); sq += __shfl_xor(sq, off); }
    float mu = s * (1.f / C);
    float var = sq * (1.f / C) - mu * mu;
    float rs = rsqrtf(var + EPSLN);
    unsigned short* yr = y + t * C;
#pragma unroll
    for (int i = 0; i < 6; ++i) {
        int c = lane + 32 * i;
        yr[c] = f2bf((v[i] - mu) * rs * g[c] + b[c]);
    }
}

// ---------------- dynamic position bias MLP ----------------
__device__ void ln_relu12(float* t, const float* g, const float* b) {
    float s = 0.f;
#pragma unroll
    for (int i = 0; i < 12; ++i) s += t[i];
    float mu = s * (1.f / 12.f), q = 0.f;
#pragma unroll
    for (int i = 0; i < 12; ++i) { float d = t[i] - mu; q += d * d; }
    float rs = rsqrtf(q * (1.f / 12.f) + EPSLN);
#pragma unroll
    for (int i = 0; i < 12; ++i) {
        float v = (t[i] - mu) * rs * g[i] + b[i];
        t[i] = v > 0.f ? v : 0.f;
    }
}

__global__ void pos_mlp_kernel(const float* pw, const float* pb,
                               const float* g1, const float* b1, const float* w1, const float* bb1,
                               const float* g2, const float* b2, const float* w2, const float* bb2,
                               const float* g3, const float* b3, const float* w3, const float* bb3,
                               float* __restrict__ tout) {
    int r = threadIdx.x;
    if (r >= 225) return;
    float dy = (float)(r / 15 - 7), dx = (float)(r % 15 - 7);
    float t[12], t2[12];
#pragma unroll
    for (int j = 0; j < 12; ++j) t[j] = dy * pw[j] + dx * pw[12 + j] + pb[j];
    ln_relu12(t, g1, b1);
    for (int j = 0; j < 12; ++j) { float s = bb1[j]; for (int i = 0; i < 12; ++i) s += t[i] * w1[i * 12 + j]; t2[j] = s; }
    ln_relu12(t2, g2, b2);
    for (int j = 0; j < 12; ++j) { float s = bb2[j]; for (int i = 0; i < 12; ++i) s += t2[i] * w2[i * 12 + j]; t[j] = s; }
    ln_relu12(t, g3, b3);
    for (int h = 0; h < HEADS; ++h) { float s = bb3[h]; for (int i = 0; i < 12; ++i) s += t[i] * w3[i * HEADS + h]; tout[r * HEADS + h] = s; }
}

__global__ void pos_gather_kernel(const float* __restrict__ tpos, float* __restrict__ bias) {
    for (int idx = blockIdx.x * blockDim.x + threadIdx.x; idx < HEADS * N * N; idx += gridDim.x * blockDim.x) {
        int h = idx >> 12, rem = idx & 4095, i = rem >> 6, j = rem & 63;
        int dy = (i >> 3) - (j >> 3), dx = (i & 7) - (j & 7);
        int id = (dy + G - 1) * (2 * G - 1) + (dx + G - 1);
        bias[idx] = tpos[id * HEADS + h];
    }
}

// ---------------- tiled WMMA GEMM: out[T,Nc] = act(A[T,K] @ W[K,Nc] + bias) (+resid) ----------------
// W supplied pre-transposed as Wt[Nc][K] so both tiles stage via async b128 copies.
// Block tile 128x64, 8 waves, 32x32 per wave (2x2 WMMA 16x16 tiles), K-step 32,
// double-buffered LDS with async global->LDS staging overlapped with WMMA.
// MODE 0: store bf16. MODE 1: store f32 with residual add.
template<bool GELU, int MODE>
__global__ __launch_bounds__(256)
void gemm_bf16_kernel(const unsigned short* __restrict__ A, const unsigned short* __restrict__ Wt,
                      const float* __restrict__ bias, const float* __restrict__ resid,
                      void* __restrict__ outp, int K, int Nc) {
    __shared__ unsigned short lA[2][128 * 32];
    __shared__ unsigned short lB[2][64 * 32];   // lB[n][k]
    const int m0 = blockIdx.x * 128, n0 = blockIdx.y * 64;
    const int tid = threadIdx.x;
    const int wid = tid >> 5, wm = wid & 3, wn = wid >> 2;

    v8f acc[2][2];
#pragma unroll
    for (int i = 0; i < 2; ++i)
#pragma unroll
        for (int j = 0; j < 2; ++j)
            acc[i][j] = (v8f){0.f, 0.f, 0.f, 0.f, 0.f, 0.f, 0.f, 0.f};

    const int arow = tid >> 1, aseg = (tid & 1) * 16;   // A: 128 rows, 2x8-elem segs per thread
    const int brow = tid >> 2, bseg = (tid & 3) * 8;    // B: 64 rows,  1x8-elem  seg per thread

    const unsigned short* Abase = A  + (size_t)(m0 + arow) * K + aseg;
    const unsigned short* Bbase = Wt + (size_t)(n0 + brow) * K + bseg;

    auto stage = [&](int k0, int bufi) {
        async_copy_b128(&lA[bufi][arow * 32 + aseg],     Abase + k0);
        async_copy_b128(&lA[bufi][arow * 32 + aseg + 8], Abase + k0 + 8);
        async_copy_b128(&lB[bufi][brow * 32 + bseg],     Bbase + k0);
    };

    const int kIters = K >> 5;
    stage(0, 0);
    for (int it = 0; it < kIters; ++it) {
        wait_async0();           // own async stores to LDS complete
        __syncthreads();         // everyone's tile for `it` visible
        if (it + 1 < kIters) stage((it + 1) << 5, (it + 1) & 1);

        const unsigned short* cA = lA[it & 1];
        const unsigned short* cB = lB[it & 1];
        Frag a0 = load_frag(cA, 32, wm * 32 + 0);
        Frag a1 = load_frag(cA, 32, wm * 32 + 16);
        Frag b0 = load_frag(cB, 32, wn * 32 + 0);
        Frag b1 = load_frag(cB, 32, wn * 32 + 16);
        acc[0][0] = wmma_bf16(a0, b0, acc[0][0]);
        acc[0][1] = wmma_bf16(a0, b1, acc[0][1]);
        acc[1][0] = wmma_bf16(a1, b0, acc[1][0]);
        acc[1][1] = wmma_bf16(a1, b1, acc[1][1]);
        __syncthreads();         // done reading buf before it is restaged at it+2
    }

    // epilogue
    const int lane = tid & 31, nl = lane & 15, kh = lane >> 4;
    float* fout = (float*)outp;
    unsigned short* bout = (unsigned short*)outp;
#pragma unroll
    for (int mi = 0; mi < 2; ++mi)
#pragma unroll
        for (int nj = 0; nj < 2; ++nj) {
            int mb = m0 + wm * 32 + mi * 16;
            int col = n0 + wn * 32 + nj * 16 + nl;
            float bv = bias[col];
#pragma unroll
            for (int r = 0; r < 8; ++r) {
                int row = mb + r + (kh << 3);
                float v = acc[mi][nj][r] + bv;
                if (GELU) v = 0.5f * v * (1.f + erff(v * 0.70710678118654752f));
                size_t o = (size_t)row * Nc + col;
                if (MODE == 1) { fout[o] = v + resid[o]; }
                else           { bout[o] = f2bf(v); }
            }
        }
}

// ---------------- windowed attention: one block per (window, head), 2 waves ----------------
__global__ __launch_bounds__(64)
void attn_kernel(const unsigned short* __restrict__ qkv, const float* __restrict__ bias,
                 unsigned short* __restrict__ o) {
    __shared__ unsigned short lq[N * DH];    // q rows [tok][d]
    __shared__ unsigned short lk[N * DH];    // k rows [tok][d]
    __shared__ unsigned short lvT[DH * N];   // v transposed [d][tok]
    __shared__ unsigned short lp[N * N];     // softmax probs bf16 [m][n]

    const int w = blockIdx.x, h = blockIdx.y;
    const int bimg = w >> 8, wi = w & 255, wy = wi >> 4, wx = wi & 15;
    const int tid = threadIdx.x;
    const int tbase = bimg * (HH * WWd) + (wy * G) * WWd + wx * G;

    // q/k via async global->LDS; v loaded to regs and scattered transposed
    for (int idx = tid; idx < N * 16; idx += 64) {
        int tok = idx >> 4, c2 = idx & 15;
        int tg = tbase + (tok >> 3) * WWd + (tok & 7);
        const unsigned short* row = qkv + (size_t)tg * (3 * C);
        async_copy_b32(&lq[tok * 32 + c2 * 2], row + h * 32 + c2 * 2);
        async_copy_b32(&lk[tok * 32 + c2 * 2], row + 192 + h * 32 + c2 * 2);
        uint32_t vv = *(const uint32_t*)(row + 384 + h * 32 + c2 * 2);
        lvT[(c2 * 2) * N + tok]     = (unsigned short)(vv & 0xffffu);
        lvT[(c2 * 2 + 1) * N + tok] = (unsigned short)(vv >> 16);
    }
    wait_async0();
    __syncthreads();

    const int wv = tid >> 5, lane = tid & 31, nl = lane & 15, kh = lane >> 4;
    const float scale = 0.17677669529663687f;   // 1/sqrt(32)
    const float* bh = bias + (size_t)h * (N * N);

    // S = (q @ k^T) * scale + bias   (wave wv owns rows [wv*32, wv*32+32))
    Frag qa[2], kb[4];
#pragma unroll
    for (int mi = 0; mi < 2; ++mi) qa[mi] = load_frag(lq, DH, wv * 32 + mi * 16);
#pragma unroll
    for (int nj = 0; nj < 4; ++nj) kb[nj] = load_frag(lk, DH, nj * 16);

    float sv[2][4][8];
#pragma unroll
    for (int mi = 0; mi < 2; ++mi)
#pragma unroll
        for (int nj = 0; nj < 4; ++nj) {
            v8f z = (v8f){0.f, 0.f, 0.f, 0.f, 0.f, 0.f, 0.f, 0.f};
            v8f s = wmma_bf16(qa[mi], kb[nj], z);
#pragma unroll
            for (int r = 0; r < 8; ++r) {
                int m = wv * 32 + mi * 16 + r + (kh << 3);
                sv[mi][nj][r] = s[r] * scale + bh[m * N + nj * 16 + nl];
            }
        }

    // row softmax (row spread over 4 N-tiles x 16 lanes within each half)
#pragma unroll
    for (int mi = 0; mi < 2; ++mi)
#pragma unroll
        for (int r = 0; r < 8; ++r) {
            float mx = sv[mi][0][r];
#pragma unroll
            for (int nj = 1; nj < 4; ++nj) mx = fmaxf(mx, sv[mi][nj][r]);
#pragma unroll
            for (int off = 1; off < 16; off <<= 1) mx = fmaxf(mx, __shfl_xor(mx, off));
            float e[4]; float sum = 0.f;
#pragma unroll
            for (int nj = 0; nj < 4; ++nj) { e[nj] = expf(sv[mi][nj][r] - mx); sum += e[nj]; }
#pragma unroll
            for (int off = 1; off < 16; off <<= 1) sum += __shfl_xor(sum, off);
            float inv = 1.f / sum;
            int m = wv * 32 + mi * 16 + r + (kh << 3);
#pragma unroll
            for (int nj = 0; nj < 4; ++nj) lp[m * N + nj * 16 + nl] = f2bf(e[nj] * inv);
        }
    // same-wave LDS producer/consumer: DS ops stay in order per wave

    // O = P @ V
#pragma unroll
    for (int mi = 0; mi < 2; ++mi)
#pragma unroll
        for (int nd = 0; nd < 2; ++nd) {
            v8f a = (v8f){0.f, 0.f, 0.f, 0.f, 0.f, 0.f, 0.f, 0.f};
#pragma unroll
            for (int kc = 0; kc < 2; ++kc) {
                Frag pa = load_frag(lp + kc * 32, N, wv * 32 + mi * 16);
                Frag vb = load_frag(lvT + kc * 32, N, nd * 16);
                a = wmma_bf16(pa, vb, a);
            }
#pragma unroll
            for (int r = 0; r < 8; ++r) {
                int m = wv * 32 + mi * 16 + r + (kh << 3);
                int tg = tbase + (m >> 3) * WWd + (m & 7);
                o[(size_t)tg * C + h * DH + nd * 16 + nl] = f2bf(a[r]);
            }
        }
}

// ---------------- host side ----------------
extern "C" void kernel_launch(void* const* d_in, const int* in_sizes, int n_in,
                              void* d_out, int out_size, void* d_ws, size_t ws_size,
                              hipStream_t stream) {
    (void)in_sizes; (void)n_in; (void)out_size; (void)ws_size;
    const float* x       = (const float*)d_in[0];
    const float* y       = (const float*)d_in[1];
    const float* norm1_g = (const float*)d_in[4];
    const float* norm1_b = (const float*)d_in[5];
    const float* qkv_w   = (const float*)d_in[6];
    const float* qkv_b   = (const float*)d_in[7];
    const float* proj_w  = (const float*)d_in[8];
    const float* proj_b  = (const float*)d_in[9];
    const float* posp_w  = (const float*)d_in[10];
    const float* posp_b  = (const float*)d_in[11];
    const float* p1lg = (const float*)d_in[12]; const float* p1lb = (const float*)d_in[13];
    const float* p1w  = (const float*)d_in[14]; const float* p1b  = (const float*)d_in[15];
    const float* p2lg = (const float*)d_in[16]; const float* p2lb = (const float*)d_in[17];
    const float* p2w  = (const float*)d_in[18]; const float* p2b  = (const float*)d_in[19];
    const float* p3lg = (const float*)d_in[20]; const float* p3lb = (const float*)d_in[21];
    const float* p3w  = (const float*)d_in[22]; const float* p3b  = (const float*)d_in[23];
    const float* norm2_g = (const float*)d_in[24];
    const float* norm2_b = (const float*)d_in[25];
    const float* m1w_f[2] = { (const float*)d_in[26], (const float*)d_in[30] };
    const float* m1b_f[2] = { (const float*)d_in[27], (const float*)d_in[31] };
    const float* m2w_f[2] = { (const float*)d_in[28], (const float*)d_in[32] };
    const float* m2b_f[2] = { (const float*)d_in[29], (const float*)d_in[33] };

    // workspace carve-out
    size_t off = 0;
    auto alloc = [&](size_t bytes) -> void* {
        off = (off + 255) & ~(size_t)255;
        void* p = (char*)d_ws + off;
        off += bytes;
        return p;
    };
    float* tpos   = (float*)alloc(225 * HEADS * sizeof(float));
    float* biasb  = (float*)alloc((size_t)HEADS * N * N * sizeof(float));
    unsigned short* wqkv  = (unsigned short*)alloc((size_t)C * 3 * C * 2);  // [576][192]
    unsigned short* wproj = (unsigned short*)alloc((size_t)C * C * 2);      // [192][192]
    unsigned short* wm1[2], * wm2[2];
    for (int i = 0; i < 2; ++i) {
        wm1[i] = (unsigned short*)alloc((size_t)C * 4 * C * 2);             // [768][192]
        wm2[i] = (unsigned short*)alloc((size_t)4 * C * C * 2);             // [192][768]
    }
    unsigned short* bufA   = (unsigned short*)alloc((size_t)T_TOK * C * 2);       // xn / h
    unsigned short* bufO   = (unsigned short*)alloc((size_t)T_TOK * C * 2);       // attn out
    unsigned short* bufBig = (unsigned short*)alloc((size_t)T_TOK * 4 * C * 2);   // qkv / gelu(mlp1)

    // weight conversion (transposed: Wt[N][K])
    auto cvtT = [&](const float* in, unsigned short* out, int K, int Nc) {
        int n = K * Nc;
        cvt_transpose_kernel<<<(n + 255) / 256, 256, 0, stream>>>(in, out, K, Nc);
    };
    cvtT(qkv_w, wqkv, C, 3 * C);
    cvtT(proj_w, wproj, C, C);
    for (int i = 0; i < 2; ++i) {
        cvtT(m1w_f[i], wm1[i], C, 4 * C);
        cvtT(m2w_f[i], wm2[i], 4 * C, C);
    }

    // position bias
    pos_mlp_kernel<<<1, 256, 0, stream>>>(posp_w, posp_b, p1lg, p1lb, p1w, p1b,
                                          p2lg, p2lb, p2w, p2b, p3lg, p3lb, p3w, p3b, tpos);
    pos_gather_kernel<<<96, 256, 0, stream>>>(tpos, biasb);

    const dim3 gQKV(T_TOK / 128, (3 * C) / 64);
    const dim3 gPRJ(T_TOK / 128, C / 64);
    const dim3 gM1 (T_TOK / 128, (4 * C) / 64);
    const dim3 gATT(Bb * (HH / G) * (WWd / G), HEADS);

    for (int br = 0; br < 2; ++br) {
        const float* inp = br == 0 ? x : y;
        float* outb = (float*)d_out + (size_t)br * T_TOK * C;

        ln_kernel<<<T_TOK / 8, 256, 0, stream>>>(inp, norm1_g, norm1_b, bufA);
        gemm_bf16_kernel<false, 0><<<gQKV, 256, 0, stream>>>(bufA, wqkv, qkv_b, nullptr, bufBig, C, 3 * C);
        attn_kernel<<<gATT, 64, 0, stream>>>(bufBig, biasb, bufO);
        gemm_bf16_kernel<false, 1><<<gPRJ, 256, 0, stream>>>(bufO, wproj, proj_b, inp, outb, C, C);
        ln_kernel<<<T_TOK / 8, 256, 0, stream>>>(outb, norm2_g, norm2_b, bufA);
        gemm_bf16_kernel<true, 0><<<gM1, 256, 0, stream>>>(bufA, wm1[br], m1b_f[br], nullptr, bufBig, C, 4 * C);
        gemm_bf16_kernel<false, 1><<<gPRJ, 256, 0, stream>>>(bufBig, wm2[br], m2b_f[br], outb, outb, 4 * C, C);
    }
}
